// speller_66142496358839
// MI455X (gfx1250) — matrix operand order repaired and checked by reading, not compiled, not run on hardware
//
#include <hip/hip_runtime.h>
#include <math.h>

// ---------------------------------------------------------------- types
typedef __attribute__((ext_vector_type(16))) __bf16 v16bf;
typedef __attribute__((ext_vector_type(8)))  float  v8f;

#define B_    32
#define T_    2048
#define L_    100
#define HID_  512
#define E_    64
#define V_    64
#define G_    2048      // 4*HID
#define IN0_  576       // E + HID
#define STEPS_ 99
#define NBLK_ 16        // cooperating workgroups in the recurrent scan (== max cluster size)

// ---------------------------------------------------------------- helpers
__device__ __forceinline__ float sigmoidf_(float x) {
    return 1.0f / (1.0f + __expf(-x));
}

__device__ __forceinline__ float wave_reduce_sum(float v) {
    #pragma unroll
    for (int off = 16; off > 0; off >>= 1)
        v += __shfl_xor(v, off, 32);
    return v;
}

// Cross-workgroup sync for the 16-block recurrent scan.
// Fast path: hardware cluster barrier (s_barrier_signal/-wait -3) — NOP if the
// dispatch is not clustered (ISA 3.2). Correctness path: monotonic L2 atomic
// counter (no reset races; counter zeroed by a tiny kernel before each scan).
__device__ __forceinline__ void grid_sync(int* cnt, int phase) {
    __threadfence();
    __syncthreads();
    if (threadIdx.x == 0) {
        __builtin_amdgcn_s_cluster_barrier();
        __hip_atomic_fetch_add(cnt, 1, __ATOMIC_RELEASE, __HIP_MEMORY_SCOPE_AGENT);
        while (__hip_atomic_load(cnt, __ATOMIC_ACQUIRE, __HIP_MEMORY_SCOPE_AGENT) < NBLK_ * phase)
            __builtin_amdgcn_s_sleep(1);
    }
    __syncthreads();
}

// ---------------------------------------------------------------- fp32 -> bf16 convert
__global__ void k_cvt_bf16(const float* __restrict__ src, __bf16* __restrict__ dst, int n) {
    int i = blockIdx.x * blockDim.x + threadIdx.x;
    if (i < n) dst[i] = (__bf16)src[i];
}

__global__ void k_zero_sync(int* c) { if (threadIdx.x == 0) *c = 0; }

// ---------------------------------------------------------------- init: rnn_in0 = [one_hot(gt[:,0]) | context[:,0,:]], h=c=0
__global__ void k_init(const float* __restrict__ context, const int* __restrict__ gt,
                       float* __restrict__ rnn_in, float* __restrict__ h, float* __restrict__ c) {
    int b = blockIdx.x;           // 32 blocks
    int j = threadIdx.x;          // 576 threads
    if (b == 0 && j < HID_) {
        #pragma unroll
        for (int l = 0; l < 3; ++l) { h[l * HID_ + j] = 0.0f; c[l * HID_ + j] = 0.0f; }
    }
    if (j < E_) {
        int tok = gt[b * L_ + 0];
        rnn_in[b * IN0_ + j] = (j == tok) ? 1.0f : 0.0f;   // one-hot seed (V == E == 64)
    } else {
        rnn_in[b * IN0_ + j] = context[(size_t)b * T_ * HID_ + (j - E_)];
    }
}

// ---------------------------------------------------------------- rnn_in fp32 -> bf16
__global__ void k_cvt_in(const float* __restrict__ src, __bf16* __restrict__ dst) {
    int i = blockIdx.x * blockDim.x + threadIdx.x;
    if (i < B_ * IN0_) dst[i] = (__bf16)src[i];
}

// ---------------------------------------------------------------- WMMA GEMM: Xg[32,2048] = rnn_in_bf[32,576] @ w_ih0_bf^T
// grid = 256 blocks x 32 threads (one wave per 16x16 output tile)
__global__ void k_gate_x(const __bf16* __restrict__ A,   // [32][576]
                         const __bf16* __restrict__ W,   // [2048][576] (row n holds k=0..575)
                         float* __restrict__ Xg) {       // [32][2048]
    const int tile = blockIdx.x;          // 0..255
    const int nt   = tile & 127;          // 128 tiles along gates
    const int mt   = tile >> 7;           // 2 tiles along batch
    const int m0   = mt * 16;
    const int n0   = nt * 16;
    const int lane = threadIdx.x;         // wave32, all lanes active (EXEC all 1s)
    const int lo16 = lane & 15;
    const int hi   = lane >> 4;           // 0 or 1

    v8f acc = {};
    const int arow   = m0 + lo16;         // A fragment row for this lane
    const int bcol   = n0 + lo16;         // B fragment column for this lane
    const int khalfA = hi * 16;           // A: K = khalfA + e (0..15 or 16..31)
    const int kbaseB = hi * 16;           // B: K = kbaseB + e

    for (int kt = 0; kt < IN0_ / 32; ++kt) {     // 18 k-slices of 32
        const int k0 = kt * 32;
        v16bf a, b;
        #pragma unroll
        for (int e = 0; e < 8; ++e) {
            a[e]     = A[arow * IN0_ + k0 + khalfA + e];
            a[e + 8] = A[arow * IN0_ + k0 + khalfA + 8 + e];
        }
        #pragma unroll
        for (int e = 0; e < 16; ++e) {
            b[e] = W[bcol * IN0_ + k0 + kbaseB + e];   // B(k,n) = w_ih0[n][k]
        }
        acc = __builtin_amdgcn_wmma_f32_16x16x32_bf16(
                  false, a, false, b, (short)0, acc, false, false);
    }

    const int n = n0 + lo16;
    #pragma unroll
    for (int r = 0; r < 8; ++r) {
        const int m = m0 + r + hi * 8;    // C/D layout: VGPR r -> M = r + 8*(lane>=16)
        Xg[m * G_ + n] = acc[r];
    }
}

// ---------------------------------------------------------------- sequential 3-layer LSTM scan, 16 cooperating workgroups
// Block j owns hidden units [32j, 32j+32): its 4x32 gate rows and its c-slice are
// block-local; only the 512-float h vectors are shared through L2 + grid_sync.
__global__ void k_lstm_scan16(const float* __restrict__ Xg,
                              const __bf16* __restrict__ whh0,
                              const __bf16* __restrict__ wih1, const __bf16* __restrict__ whh1,
                              const __bf16* __restrict__ wih2, const __bf16* __restrict__ whh2,
                              const float* __restrict__ bih0, const float* __restrict__ bhh0,
                              const float* __restrict__ bih1, const float* __restrict__ bhh1,
                              const float* __restrict__ bih2, const float* __restrict__ bhh2,
                              float* __restrict__ hg, float* __restrict__ cg,
                              float* __restrict__ rnn_out, int* sync_cnt) {
    __shared__ float hshA[HID_];
    __shared__ float hshB[HID_];
    __shared__ float gbuf[128];
    __shared__ float c_loc[3][32];

    const int tid   = threadIdx.x;        // 256 threads = 8 waves
    const int wave  = tid >> 5;
    const int lane  = tid & 31;
    const int unit0 = blockIdx.x * 32;
    int bar = 0;

    if (tid < 32) {
        #pragma unroll
        for (int l = 0; l < 3; ++l) c_loc[l][tid] = cg[l * HID_ + unit0 + tid];
    }

    for (int item = 0; item < B_; ++item) {
        // ------------- layer 0: gates = Xg[item] + b + w_hh0 @ h0
        for (int k = tid; k < HID_; k += 256) hshB[k] = hg[0 * HID_ + k];
        __syncthreads();
        for (int gl = wave; gl < 128; gl += 8) {          // wave-per-gate, lane-parallel K
            const int row = (gl >> 5) * HID_ + unit0 + (gl & 31);
            const __bf16* wr = whh0 + (size_t)row * HID_ + lane * 16;
            const float*  hv = &hshB[lane * 16];
            float p = 0.0f;
            #pragma unroll
            for (int e = 0; e < 16; ++e) p += (float)wr[e] * hv[e];
            p = wave_reduce_sum(p);
            if (lane == 0)
                gbuf[gl] = Xg[item * G_ + row] + bih0[row] + bhh0[row] + p;
        }
        __syncthreads();
        if (tid < 32) {
            float i = sigmoidf_(gbuf[tid]);
            float f = sigmoidf_(gbuf[32 + tid]);
            float g = tanhf(gbuf[64 + tid]);
            float o = sigmoidf_(gbuf[96 + tid]);
            float cn = f * c_loc[0][tid] + i * g;
            c_loc[0][tid] = cn;
            hg[0 * HID_ + unit0 + tid] = o * tanhf(cn);
        }
        grid_sync(sync_cnt, ++bar);

        // ------------- layer 1: gates = b + w_ih1 @ h0_new + w_hh1 @ h1
        for (int k = tid; k < HID_; k += 256) {
            hshA[k] = hg[0 * HID_ + k];
            hshB[k] = hg[1 * HID_ + k];
        }
        __syncthreads();
        for (int gl = wave; gl < 128; gl += 8) {
            const int row = (gl >> 5) * HID_ + unit0 + (gl & 31);
            const __bf16* wa = wih1 + (size_t)row * HID_ + lane * 16;
            const __bf16* wb = whh1 + (size_t)row * HID_ + lane * 16;
            const float*  ha = &hshA[lane * 16];
            const float*  hb = &hshB[lane * 16];
            float p = 0.0f;
            #pragma unroll
            for (int e = 0; e < 16; ++e) p += (float)wa[e] * ha[e] + (float)wb[e] * hb[e];
            p = wave_reduce_sum(p);
            if (lane == 0) gbuf[gl] = bih1[row] + bhh1[row] + p;
        }
        __syncthreads();
        if (tid < 32) {
            float i = sigmoidf_(gbuf[tid]);
            float f = sigmoidf_(gbuf[32 + tid]);
            float g = tanhf(gbuf[64 + tid]);
            float o = sigmoidf_(gbuf[96 + tid]);
            float cn = f * c_loc[1][tid] + i * g;
            c_loc[1][tid] = cn;
            hg[1 * HID_ + unit0 + tid] = o * tanhf(cn);
        }
        grid_sync(sync_cnt, ++bar);

        // ------------- layer 2: gates = b + w_ih2 @ h1_new + w_hh2 @ h2
        for (int k = tid; k < HID_; k += 256) {
            hshA[k] = hg[1 * HID_ + k];
            hshB[k] = hg[2 * HID_ + k];
        }
        __syncthreads();
        for (int gl = wave; gl < 128; gl += 8) {
            const int row = (gl >> 5) * HID_ + unit0 + (gl & 31);
            const __bf16* wa = wih2 + (size_t)row * HID_ + lane * 16;
            const __bf16* wb = whh2 + (size_t)row * HID_ + lane * 16;
            const float*  ha = &hshA[lane * 16];
            const float*  hb = &hshB[lane * 16];
            float p = 0.0f;
            #pragma unroll
            for (int e = 0; e < 16; ++e) p += (float)wa[e] * ha[e] + (float)wb[e] * hb[e];
            p = wave_reduce_sum(p);
            if (lane == 0) gbuf[gl] = bih2[row] + bhh2[row] + p;
        }
        __syncthreads();
        if (tid < 32) {
            float i = sigmoidf_(gbuf[tid]);
            float f = sigmoidf_(gbuf[32 + tid]);
            float g = tanhf(gbuf[64 + tid]);
            float o = sigmoidf_(gbuf[96 + tid]);
            float cn = f * c_loc[2][tid] + i * g;
            c_loc[2][tid] = cn;
            float hn = o * tanhf(cn);
            hg[2 * HID_ + unit0 + tid] = hn;
            rnn_out[item * HID_ + unit0 + tid] = hn;
        }
        grid_sync(sync_cnt, ++bar);
    }

    if (tid < 32) {
        #pragma unroll
        for (int l = 0; l < 3; ++l) cg[l * HID_ + unit0 + tid] = c_loc[l][tid];
    }
}

// ---------------------------------------------------------------- attention energy: one wave per (b,t) dot of length 512
__global__ void k_energy(const float* __restrict__ rnn_out,
                         const float* __restrict__ context,
                         float* __restrict__ att) {
    const int gwave = blockIdx.x * (blockDim.x >> 5) + (threadIdx.x >> 5);
    const int lane  = threadIdx.x & 31;
    const int b = gwave >> 11;            // 2048 t per batch item
    const int t = gwave & (T_ - 1);
    const float* crow = context + ((size_t)b * T_ + t) * HID_;
    float p = 0.0f;
    #pragma unroll 4
    for (int k = lane; k < HID_; k += 32)
        p += rnn_out[b * HID_ + k] * crow[k];
    p = wave_reduce_sum(p);
    if (lane == 0) att[b * T_ + t] = p;
}

// ---------------------------------------------------------------- softmax over T per batch item (in place)
__global__ void k_softmax(float* __restrict__ att) {
    __shared__ float red[256];
    const int b = blockIdx.x, tid = threadIdx.x;
    float* row = att + (size_t)b * T_;

    float m = -3.4e38f;
    for (int t = tid; t < T_; t += 256) m = fmaxf(m, row[t]);
    red[tid] = m; __syncthreads();
    for (int s = 128; s > 0; s >>= 1) { if (tid < s) red[tid] = fmaxf(red[tid], red[tid + s]); __syncthreads(); }
    const float rmax = red[0]; __syncthreads();

    float sum = 0.0f;
    for (int t = tid; t < T_; t += 256) {
        float e = __expf(row[t] - rmax);
        row[t] = e;
        sum += e;
    }
    red[tid] = sum; __syncthreads();
    for (int s = 128; s > 0; s >>= 1) { if (tid < s) red[tid] += red[tid + s]; __syncthreads(); }
    const float inv = 1.0f / red[0]; __syncthreads();

    for (int t = tid; t < T_; t += 256) row[t] *= inv;
}

// ---------------------------------------------------------------- ctx[b,d] = sum_t att[b,t] * context[b,t,d]
__global__ void k_ctx(const float* __restrict__ att,
                      const float* __restrict__ context,
                      float* __restrict__ ctxv) {
    const int b = blockIdx.x, d = threadIdx.x;   // 32 blocks x 512 threads
    const float* cbase = context + (size_t)b * T_ * HID_;
    float acc = 0.0f;
    for (int t = 0; t < T_; ++t) {
        __builtin_prefetch(cbase + (size_t)(t + 4) * HID_ + d, 0, 1);  // global_prefetch_b8
        acc += att[b * T_ + t] * cbase[(size_t)t * HID_ + d];
    }
    ctxv[b * HID_ + d] = acc;
}

// ---------------------------------------------------------------- output head: logits -> log_softmax -> preds; build next rnn_in
__global__ void k_output(const float* __restrict__ rnn_out, const float* __restrict__ ctxv,
                         const float* __restrict__ w_out, const float* __restrict__ b_out,
                         const int* __restrict__ gt, const float* __restrict__ embedding,
                         float* __restrict__ preds, float* __restrict__ rnn_in, int s) {
    __shared__ float logits[V_];
    __shared__ float s_max, s_lse;
    const int b = blockIdx.x, tid = threadIdx.x;   // 32 blocks x 256 threads

    if (tid < V_) {
        float acc = b_out[tid];
        const float* wr = w_out + tid * (2 * HID_);
        for (int k = 0; k < HID_; ++k) acc += wr[k]        * rnn_out[b * HID_ + k];
        for (int k = 0; k < HID_; ++k) acc += wr[HID_ + k] * ctxv[b * HID_ + k];
        logits[tid] = acc;
    }
    __syncthreads();
    if (tid == 0) {
        float m = -3.4e38f;
        for (int v = 0; v < V_; ++v) m = fmaxf(m, logits[v]);
        float sum = 0.0f;
        for (int v = 0; v < V_; ++v) sum += __expf(logits[v] - m);
        s_max = m; s_lse = __logf(sum);
    }
    __syncthreads();
    if (tid < V_)
        preds[((size_t)b * STEPS_ + s) * V_ + tid] = logits[tid] - s_max - s_lse;

    // next teacher-forced input: [embedding[gt[:, s+1]] | ctx]
    const int tok = gt[b * L_ + (s + 1)];
    if (tid < E_)
        rnn_in[b * IN0_ + tid] = embedding[tok * E_ + tid];
    for (int d = tid; d < HID_; d += 256)
        rnn_in[b * IN0_ + E_ + d] = ctxv[b * HID_ + d];
}

// ---------------------------------------------------------------- launcher
extern "C" void kernel_launch(void* const* d_in, const int* in_sizes, int n_in,
                              void* d_out, int out_size, void* d_ws, size_t ws_size,
                              hipStream_t stream) {
    const float* context   = (const float*)d_in[0];
    const int*   gt        = (const int*)  d_in[1];
    // d_in[2] = tf_rate (always 1, ignored)
    const float* embedding = (const float*)d_in[3];
    const float* w_out     = (const float*)d_in[4];
    const float* b_out     = (const float*)d_in[5];
    const float* w_ih0 = (const float*)d_in[6];
    const float* w_hh0 = (const float*)d_in[7];
    const float* b_ih0 = (const float*)d_in[8];
    const float* b_hh0 = (const float*)d_in[9];
    const float* w_ih1 = (const float*)d_in[10];
    const float* w_hh1 = (const float*)d_in[11];
    const float* b_ih1 = (const float*)d_in[12];
    const float* b_hh1 = (const float*)d_in[13];
    const float* w_ih2 = (const float*)d_in[14];
    const float* w_hh2 = (const float*)d_in[15];
    const float* b_ih2 = (const float*)d_in[16];
    const float* b_hh2 = (const float*)d_in[17];
    float* preds = (float*)d_out;

    // ---- workspace carve-up
    char* p = (char*)d_ws;
    __bf16* wih0_bf = (__bf16*)p;  p += (size_t)G_ * IN0_  * sizeof(__bf16);
    __bf16* whh0_bf = (__bf16*)p;  p += (size_t)G_ * HID_  * sizeof(__bf16);
    __bf16* wih1_bf = (__bf16*)p;  p += (size_t)G_ * HID_  * sizeof(__bf16);
    __bf16* whh1_bf = (__bf16*)p;  p += (size_t)G_ * HID_  * sizeof(__bf16);
    __bf16* wih2_bf = (__bf16*)p;  p += (size_t)G_ * HID_  * sizeof(__bf16);
    __bf16* whh2_bf = (__bf16*)p;  p += (size_t)G_ * HID_  * sizeof(__bf16);
    __bf16* rnnin_bf = (__bf16*)p; p += (size_t)B_ * IN0_  * sizeof(__bf16);
    float* rnn_in  = (float*)p;    p += (size_t)B_ * IN0_  * sizeof(float);
    float* Xg      = (float*)p;    p += (size_t)B_ * G_    * sizeof(float);
    float* hg      = (float*)p;    p += (size_t)3  * HID_  * sizeof(float);
    float* cg      = (float*)p;    p += (size_t)3  * HID_  * sizeof(float);
    float* rnn_out = (float*)p;    p += (size_t)B_ * HID_  * sizeof(float);
    float* att     = (float*)p;    p += (size_t)B_ * T_    * sizeof(float);
    float* ctxv    = (float*)p;    p += (size_t)B_ * HID_  * sizeof(float);
    int*   sync_cnt = (int*)p;     p += 64;

    // ---- once per launch: bf16 weights (WMMA projection + recurrent GEMVs)
    {
        int n0 = G_ * IN0_, n1 = G_ * HID_;
        k_cvt_bf16<<<(n0 + 255) / 256, 256, 0, stream>>>(w_ih0, wih0_bf, n0);
        k_cvt_bf16<<<(n1 + 255) / 256, 256, 0, stream>>>(w_hh0, whh0_bf, n1);
        k_cvt_bf16<<<(n1 + 255) / 256, 256, 0, stream>>>(w_ih1, wih1_bf, n1);
        k_cvt_bf16<<<(n1 + 255) / 256, 256, 0, stream>>>(w_hh1, whh1_bf, n1);
        k_cvt_bf16<<<(n1 + 255) / 256, 256, 0, stream>>>(w_ih2, wih2_bf, n1);
        k_cvt_bf16<<<(n1 + 255) / 256, 256, 0, stream>>>(w_hh2, whh2_bf, n1);
    }
    k_init<<<B_, IN0_, 0, stream>>>(context, gt, rnn_in, hg, cg);

    for (int s = 0; s < STEPS_; ++s) {
        k_cvt_in<<<(B_ * IN0_ + 255) / 256, 256, 0, stream>>>(rnn_in, rnnin_bf);
        k_gate_x<<<256, 32, 0, stream>>>(rnnin_bf, wih0_bf, Xg);
        k_zero_sync<<<1, 1, 0, stream>>>(sync_cnt);
        k_lstm_scan16<<<NBLK_, 256, 0, stream>>>(Xg, whh0_bf, wih1_bf, whh1_bf,
                                                 wih2_bf, whh2_bf,
                                                 b_ih0, b_hh0, b_ih1, b_hh1, b_ih2, b_hh2,
                                                 hg, cg, rnn_out, sync_cnt);
        k_energy<<<(B_ * T_) / 8, 256, 0, stream>>>(rnn_out, context, att);
        k_softmax<<<B_, 256, 0, stream>>>(att);
        k_ctx<<<B_, HID_, 0, stream>>>(att, context, ctxv);
        k_output<<<B_, 256, 0, stream>>>(rnn_out, ctxv, w_out, b_out, gt, embedding,
                                         preds, rnn_in, s);
    }
}
